// NegSamplingWord2Vec_3375844295074
// MI455X (gfx1250) — matrix operand order repaired and checked by reading, not compiled, not run on hardware
//
#include <hip/hip_runtime.h>

// ---- CDNA5 WMMA types (probe-confirmed signatures) ----
typedef __attribute__((ext_vector_type(16))) __bf16 v16bf;
typedef __attribute__((ext_vector_type(8)))  float  v8f;

#define WORD_DIM   128
#define W_CTX      10
#define K_NEG      10
#define SLOTS      110     // 10 positives + 100 negatives per batch element
#define SLOTS_PW   11      // slots handled per wave
#define CHUNKS     10      // SLOTS / SLOTS_PW
#define TILE_B     16      // batch elements per wave (WMMA N dimension)
#define WPB        8       // waves per block (256 threads, wave32)

__device__ __forceinline__ float4 ld4(const float* p) {
  return *reinterpret_cast<const float4*>(p);
}

// Build one lane's v16bf A/B fragment from 16 f32 values (4x float4), cvt f32->bf16.
__device__ __forceinline__ v16bf pack16(float4 a, float4 b, float4 c, float4 d) {
  v16bf v;
  v[0]  = (__bf16)a.x; v[1]  = (__bf16)a.y; v[2]  = (__bf16)a.z; v[3]  = (__bf16)a.w;
  v[4]  = (__bf16)b.x; v[5]  = (__bf16)b.y; v[6]  = (__bf16)b.z; v[7]  = (__bf16)b.w;
  v[8]  = (__bf16)c.x; v[9]  = (__bf16)c.y; v[10] = (__bf16)c.z; v[11] = (__bf16)c.w;
  v[12] = (__bf16)d.x; v[13] = (__bf16)d.y; v[14] = (__bf16)d.z; v[15] = (__bf16)d.w;
  return v;
}

// -log_sigmoid(x) = max(-x, 0) + log(1 + exp(-|x|))   (numerically stable)
__device__ __forceinline__ float neg_log_sigmoid(float x) {
  return fmaxf(-x, 0.0f) + __logf(1.0f + __expf(-fabsf(x)));
}

__global__ void zero_out_kernel(float* __restrict__ out, int n) {
  int i = blockIdx.x * blockDim.x + threadIdx.x;
  if (i < n) out[i] = 0.0f;
}

__global__ __launch_bounds__(256)
void w2v_negsamp_wmma(const float* __restrict__ center_vectors,
                      const float* __restrict__ outside_vectors,
                      const int*   __restrict__ center_idx,     // [B]
                      const int*   __restrict__ outside_idx,    // [B, W]
                      const int*   __restrict__ neg_idx,        // [B, W, K]
                      float*       __restrict__ out,            // [B]
                      int batch) {
  const int lane   = threadIdx.x & 31;
  const int waveId = (blockIdx.x * blockDim.x + threadIdx.x) >> 5;
  const int group  = waveId / CHUNKS;          // which 16-batch tile
  const int chunk  = waveId % CHUNKS;          // which 11-slot range
  const int base   = group * TILE_B;
  if (base >= batch) return;                   // wave-uniform guard

  const int col  = lane & 15;                  // WMMA N / M index within tile
  const int half = lane >> 4;                  // lane group (K split)

  // ---------- B matrix: bf16 centers of 16 batch elements, K=128 in 4 chunks ----------
  // B 32x16 operand, contiguous k order chosen here: lane (col=N, half h) holds 16 consecutive K at h*16.
  const int   cidx = center_idx[base + col];
  const float* crow = center_vectors + (long)cidx * WORD_DIM;
  v16bf bmat[4];
#pragma unroll
  for (int j = 0; j < 4; ++j) {
    const float* p = crow + j * 32 + half * 16;
    bmat[j] = pack16(ld4(p), ld4(p + 4), ld4(p + 8), ld4(p + 12));
  }

  // ---------- per-lane bookkeeping ----------
  // Diagonal D(i,i): i<8 -> lane i, acc[i]; i>=8 -> lane 16+i, acc[i-8].
  const bool isdiag = (lane < 8) || (lane >= 24);
  const int  r      = lane & 7;
  const int  b_lane = base + (lane & 7) + ((lane >= 16) ? 8 : 0);  // diag lane's batch elem
  const int  b_row  = base + col;                                  // A-row loader's batch elem

  unsigned maskbits = 0;
#pragma unroll
  for (int w = 0; w < W_CTX; ++w)
    if (outside_idx[b_lane * W_CTX + w] != 0) maskbits |= (1u << w);

  float lacc = 0.0f;
  const int m0 = chunk * SLOTS_PW;

  for (int m = m0; m < m0 + SLOTS_PW; ++m) {
    // gather index of slot m for this lane's A row (lane L and L+16 load the same row)
    const int vidx = (m < W_CTX) ? outside_idx[b_row * W_CTX + m]
                                 : neg_idx[b_row * (W_CTX * K_NEG) + (m - W_CTX)];
    const float* vrow = outside_vectors + (long)vidx * WORD_DIM;

    // prefetch next slot's row into cache while this slot computes
    if (m + 1 < m0 + SLOTS_PW) {
      const int nvidx = ((m + 1) < W_CTX) ? outside_idx[b_row * W_CTX + (m + 1)]
                                          : neg_idx[b_row * (W_CTX * K_NEG) + (m + 1 - W_CTX)];
      __builtin_prefetch(outside_vectors + (long)nvidx * WORD_DIM, 0, 1);
    }

    // ---------- A fragments + 4 chained WMMAs over K=128 ----------
    // A 16x32 operand, contiguous k order chosen here: lane (M=col, half h) holds 16 consecutive K at h*16 (matches B).
    v8f acc = {};
#pragma unroll
    for (int j = 0; j < 4; ++j) {
      const float* p0 = vrow + j * 32 + half * 16;
      const float* p1 = p0 + 8;
      v16bf a = pack16(ld4(p0), ld4(p0 + 4), ld4(p1), ld4(p1 + 4));
      acc = __builtin_amdgcn_wmma_f32_16x16x32_bf16(
          /*neg_a=*/false, a, /*neg_b=*/false, bmat[j],
          /*c_mod=*/(short)0, acc, /*reuse_a=*/false, /*reuse_b=*/false);
    }

    // ---------- extract diagonal value for this lane ----------
    float d = acc[0];
    d = (r == 1) ? acc[1] : d;
    d = (r == 2) ? acc[2] : d;
    d = (r == 3) ? acc[3] : d;
    d = (r == 4) ? acc[4] : d;
    d = (r == 5) ? acc[5] : d;
    d = (r == 6) ? acc[6] : d;
    d = (r == 7) ? acc[7] : d;

    // ---------- loss term (m is wave-uniform -> scalar control) ----------
    const int   w    = (m < W_CTX) ? m : ((m - W_CTX) / K_NEG);
    const float x    = (m < W_CTX) ? d : -d;          // positives: s, negatives: -s
    const float term = neg_log_sigmoid(x);
    const bool  use  = isdiag && (((maskbits >> w) & 1u) != 0u);
    lacc += use ? term : 0.0f;
  }

  if (isdiag) atomicAdd(&out[b_lane], lacc);
}

extern "C" void kernel_launch(void* const* d_in, const int* in_sizes, int n_in,
                              void* d_out, int out_size, void* d_ws, size_t ws_size,
                              hipStream_t stream) {
  const float* center_vectors  = (const float*)d_in[0];
  const float* outside_vectors = (const float*)d_in[1];
  const int*   center_idx      = (const int*)d_in[2];
  const int*   outside_idx     = (const int*)d_in[3];
  const int*   neg_idx         = (const int*)d_in[4];
  float*       out             = (float*)d_out;
  const int    batch           = in_sizes[2];  // 8192

  // d_out is poisoned by the harness; results are atomically accumulated.
  zero_out_kernel<<<(out_size + 255) / 256, 256, 0, stream>>>(out, out_size);

  const int groups = (batch + TILE_B - 1) / TILE_B;   // 512
  const int waves  = groups * CHUNKS;                 // 5120
  const int blocks = (waves + WPB - 1) / WPB;         // 640
  w2v_negsamp_wmma<<<blocks, 32 * WPB, 0, stream>>>(
      center_vectors, outside_vectors, center_idx, outside_idx, neg_idx, out, batch);
}
